// ConvCapsDR_51075751084745
// MI455X (gfx1250) — matrix operand order repaired and checked by reading, not compiled, not run on hardware
//
#include <hip/hip_runtime.h>

// ---------------------------------------------------------------------------
// Fully fused ConvCaps + dynamic routing for MI455X (gfx1250, wave32, WMMA).
//
// Shapes: x(8,8,32,32,32) f32, conv_w(16384,8,3,3) f32, prior(1,32,32,1,1),
// out s(8,16,32,30,30) f32.
//
// Never materialize u (472 MB / ~2.8 GB of HBM traffic). Per workgroup:
// one (batch, out-row h, 15-wide w tile). The x tile is DMA'd into LDS by the
// Tensor Data Mover (4D tile: 17 x 3 x 32 x 8 fp32), converted once to bf16
// WMMA B-fragments in LDS, reused across 3 routing sweeps. Each sweep streams
// over D: u_D = W_D @ patch via v_wmma_f32_16x16x32_bf16 (K=72 pad 96),
// b += sum_c u*v via LDS float atomics, per-D softmax over d (16 waves x
// 32 lanes, shfl reductions), s += cc*u in registers. Squash over c in LDS
// between sweeps. Output written straight from accumulator registers.
// ---------------------------------------------------------------------------

#define B_    8
#define CIN   8
#define DIN   32
#define COUT  16
#define DOUT  32
#define HIN   32
#define WIN   32
#define HOUT  30
#define WOUT  30
#define KK    72     // CIN*3*3
#define KSTEPS 3     // ceil(72/32) k-steps of 32
#define NP    16     // WMMA N tile (positions)
#define NPV   15     // valid positions per tile (2 tiles cover W=30)
#define XTW   17     // x-tile width (15 positions + 2 halo)

typedef __attribute__((ext_vector_type(16))) __bf16 v16bf;
typedef __attribute__((ext_vector_type(8)))  float  v8f;
typedef __attribute__((ext_vector_type(4)))  unsigned int v4u;
typedef __attribute__((ext_vector_type(8)))  int    v8i;
typedef __attribute__((ext_vector_type(4)))  int    v4i;

#if defined(__gfx1250__) && __has_builtin(__builtin_amdgcn_tensor_load_to_lds)
#define USE_TDM 1
#else
#define USE_TDM 0
#endif

// ---------------------------------------------------------------------------
// Prepack conv_w (fp32, row = group-D out channel m=c*32+d, 72 taps) into
// bf16 A-fragments: [D][mtile 0..31][ks 0..2][lane 0..31][i 0..15].
// Per ISA 16-bit A layout (16x32): lanes 0-15 hold K in {0..7, 16..23},
// lanes 16-31 hold K in {8..15, 24..31}; row M = lane%16 within the tile.
// ---------------------------------------------------------------------------
__global__ __launch_bounds__(256) void prepack_w(const float* __restrict__ w,
                                                 __bf16* __restrict__ pw,
                                                 int total) {
  int idx = blockIdx.x * 256 + threadIdx.x;
  if (idx >= total) return;
  int i    = idx & 15;
  int lane = (idx >> 4) & 31;
  int ks   = (idx >> 9) % KSTEPS;
  int mtD  = idx / (16 * 32 * KSTEPS);
  int mt   = mtD & 31;
  int D    = mtD >> 5;
  int hi   = lane >> 4;
  int m    = mt * 16 + (lane & 15);
  int k    = ks * 32 + ((i < 8 ? i : i + 8) + 8 * hi);
  float v  = 0.0f;
  if (k < KK) v = w[(D * 512 + m) * KK + k];
  pw[idx] = (__bf16)v;
}

// ---------------------------------------------------------------------------
// Main fused kernel. 512 threads = 16 waves; wave w owns c = w (2 M-tiles).
// ---------------------------------------------------------------------------
__global__ __launch_bounds__(512) void capsule_fused(
    const float* __restrict__ x, const __bf16* __restrict__ pw,
    const float* __restrict__ prior, float* __restrict__ out) {

  // LDS: 96 (patch) + 64 (b) + 32 (s/v) + 2 (cc) [+51 x-stage] = ~245 KB.
  __shared__ alignas(32) __bf16 ldsPatch[DIN * KSTEPS * 32 * 16]; // B fragments
  __shared__ float ldsB[DIN * DOUT * NP];    // routing logits b[D][d][p]
  __shared__ float ldsSV[COUT * DOUT * NP];  // s staging -> v (in place)
  __shared__ float ldsCC[DOUT * NP];         // softmax coefficients for cur D
#if USE_TDM
  __shared__ alignas(16) float ldsX[CIN * DIN * 3 * XTW]; // TDM-landed x tile
#endif

  const int tid  = threadIdx.x;
  const int wave = tid >> 5;
  const int lane = tid & 31;
  const int wt   = blockIdx.x;       // 0..1
  const int h    = blockIdx.y;       // 0..29
  const int b    = blockIdx.z;       // 0..7
  const int w0   = wt * NPV;

#if USE_TDM
  // ---- Tensor Data Mover: DMA the 4D x tile into LDS ----------------------
  // Tile (innermost->outermost): w=17, h=3, D=32, cin=8 fp32; lands
  // contiguously in ldsX as [cin][D][hh][17]. Descriptors are wave-uniform
  // (built from blockIdx + kernarg pointer). One wave issues (EXEC ignored
  // by TDM), everyone syncs after s_wait_tensorcnt.
  if (wave == 0) {
    unsigned long long gaddr =
        (unsigned long long)(uintptr_t)(x + (size_t)b * (CIN * DIN * HIN * WIN)
                                          + (size_t)h * WIN + w0);
    unsigned int laddr = (unsigned int)(uintptr_t)ldsX; // low 32 = LDS offset

    v4u g0;
    g0[0] = 1u;                                      // count=1, user mode
    g0[1] = laddr;                                   // lds_addr
    g0[2] = (unsigned int)(gaddr & 0xffffffffu);     // global_addr[31:0]
    g0[3] = (unsigned int)((gaddr >> 32) & 0x01ffffffu) | (2u << 30); // type=2

    v8i g1;
    g1[0] = (int)(2u << 16);          // data_size=4B; mask/flags/pad = 0
    g1[1] = (int)(32u << 16);         // tensor_dim0 = 32 (W)
    g1[2] = (int)(32u << 16);         // tensor_dim0 hi=0 | tensor_dim1 = 32 (H)
    g1[3] = (int)((unsigned)XTW << 16); // tensor_dim1 hi=0 | tile_dim0 = 17
    g1[4] = (int)(3u | (32u << 16));  // tile_dim1 = 3 (kh) | tile_dim2 = 32 (D)
    g1[5] = WIN;                      // tensor_dim0_stride = 32
    g1[6] = (int)((unsigned)(HIN * WIN) << 16); // dim1_stride lo16 (1024)
    g1[7] = (int)((unsigned)(HIN * WIN) >> 16); // dim1_stride hi

    v4i g2;
    g2[0] = DIN;                      // tensor_dim2 = 32
    g2[1] = CIN;                      // tensor_dim3 = 8
    g2[2] = HIN * WIN;                // tensor_dim2_stride = 1024 (D step)
    g2[3] = (int)((unsigned)CIN << 16); // stride hi=0 | tile_dim3 = 8 (cin)

    v4i g3;
    g3[0] = DIN * HIN * WIN;          // tensor_dim3_stride = 32768 (cin step)
    g3[1] = 0;                        // stride hi | tensor_dim4 = 0
    g3[2] = 0;                        // tile_dim4 = 0 (unused)
    g3[3] = 0;

    // 6-arg toolchain variant: extra int32x8 operand (ISA VADDR4 slot is
    // unused/NULL) -> pass zeros. Last arg = cpol.
    v8i g4 = {0, 0, 0, 0, 0, 0, 0, 0};
    __builtin_amdgcn_tensor_load_to_lds(g0, g1, g2, g3, g4, 0);
    __builtin_amdgcn_s_wait_tensorcnt(0);
  }
  __syncthreads();
#endif

  // ---- Build bf16 patch fragments for all 32 groups (once, reused 3x) ----
  // B layout (32x16): lanes 0-15: column N=lane, K={0..7,16..23} ; lanes 16-31:
  // column N=lane-16, K={8..15,24..31}. k = cin*9 + kh*3 + kw matches conv_w taps.
  for (int idx = tid; idx < DIN * KSTEPS * 32 * 16; idx += 512) {
    int i  = idx & 15;
    int ln = (idx >> 4) & 31;
    int ks = (idx >> 9) % KSTEPS;
    int D  = idx / (16 * 32 * KSTEPS);
    int p  = ln & 15;
    int k  = ks * 32 + ((i < 8 ? i : i + 8) + 8 * (ln >> 4));
    float v = 0.0f;
    if (k < KK && p < NPV) {
      int cin = k / 9, r = k % 9, kh = r / 3, kw = r % 3;
#if USE_TDM
      v = ldsX[((cin * DIN + D) * 3 + kh) * XTW + (p + kw)];
#else
      v = x[(((b * CIN + cin) * DIN + D) * HIN + (h + kh)) * WIN + (w0 + p + kw)];
#endif
    }
    ldsPatch[idx] = (__bf16)v;
  }
  // ---- b init from prior (broadcast over positions) ----
  for (int idx = tid; idx < DIN * DOUT * NP; idx += 512)
    ldsB[idx] = prior[idx >> 4];     // idx = (D*32+d)*16+p
  __syncthreads();

  const int p  = lane & 15;          // position within tile (WMMA N)
  const int hi = lane >> 4;          // C/D fragment: row M = j + 8*hi
  const int c  = wave;               // capsule-out channel owned by this wave

  const v8f zero8 = {0.f, 0.f, 0.f, 0.f, 0.f, 0.f, 0.f, 0.f};
  v8f sacc0 = zero8, sacc1 = zero8;  // s[c, d=0..15/16..31 slice, p]

  for (int sweep = 0; sweep < 3; ++sweep) {
    sacc0 = zero8;
    sacc1 = zero8;
    for (int D = 0; D < DIN; ++D) {
      // ---- u_D fragments: 2 M-tiles x 3 K-steps of bf16 WMMA ----
      v8f u0 = zero8, u1 = zero8;
#pragma unroll
      for (int ks = 0; ks < KSTEPS; ++ks) {
        v16bf bb = *(const v16bf*)&ldsPatch[((D * KSTEPS + ks) * 32 + lane) * 16];
        v16bf a0 = *(const v16bf*)&pw[(((D * 32 + (2 * wave + 0)) * KSTEPS + ks) * 32 + lane) * 16];
        v16bf a1 = *(const v16bf*)&pw[(((D * 32 + (2 * wave + 1)) * KSTEPS + ks) * 32 + lane) * 16];
        u0 = __builtin_amdgcn_wmma_f32_16x16x32_bf16(false, a0, false, bb,
                                                     (short)0, u0, false, false);
        u1 = __builtin_amdgcn_wmma_f32_16x16x32_bf16(false, a1, false, bb,
                                                     (short)0, u1, false, false);
      }
      // ---- deferred b-update: b[D,d,p] += sum_c u[D,c,d,p] * v[c,d,p] ----
      if (sweep > 0) {
#pragma unroll
        for (int j = 0; j < 8; ++j) {
          int d0 = j + 8 * hi;
          int d1 = 16 + j + 8 * hi;
          atomicAdd(&ldsB[(D * DOUT + d0) * NP + p],
                    u0[j] * ldsSV[(c * DOUT + d0) * NP + p]);
          atomicAdd(&ldsB[(D * DOUT + d1) * NP + p],
                    u1[j] * ldsSV[(c * DOUT + d1) * NP + p]);
        }
      }
      __syncthreads();
      // ---- softmax over d for this D, fully parallel: wave = position p,
      //      lane = d; max/sum via wave-wide shfl_xor reductions ----
      {
        float bv = ldsB[(D * DOUT + lane) * NP + wave];
        float mx = bv;
        mx = fmaxf(mx, __shfl_xor(mx, 16));
        mx = fmaxf(mx, __shfl_xor(mx, 8));
        mx = fmaxf(mx, __shfl_xor(mx, 4));
        mx = fmaxf(mx, __shfl_xor(mx, 2));
        mx = fmaxf(mx, __shfl_xor(mx, 1));
        float e = __expf(bv - mx);
        float sum = e;
        sum += __shfl_xor(sum, 16);
        sum += __shfl_xor(sum, 8);
        sum += __shfl_xor(sum, 4);
        sum += __shfl_xor(sum, 2);
        sum += __shfl_xor(sum, 1);
        ldsCC[lane * NP + wave] = e * (1.0f / sum);
      }
      __syncthreads();
      // ---- s[c,d,p] += cc[d,p] * u_D[c,d,p] (register accumulators) ----
#pragma unroll
      for (int j = 0; j < 8; ++j) {
        int d0 = j + 8 * hi;
        int d1 = 16 + j + 8 * hi;
        sacc0[j] += ldsCC[d0 * NP + p] * u0[j];
        sacc1[j] += ldsCC[d1 * NP + p] * u1[j];
      }
    } // D stream

    // ---- squash over c -> v (skip after final sweep; s is the output) ----
    if (sweep < 2) {
      __syncthreads();  // all v-reads of this sweep are done
#pragma unroll
      for (int j = 0; j < 8; ++j) {
        ldsSV[(c * DOUT + (j + 8 * hi)) * NP + p]      = sacc0[j];
        ldsSV[(c * DOUT + (16 + j + 8 * hi)) * NP + p] = sacc1[j];
      }
      __syncthreads();
      {
        int d = tid >> 4, pp = tid & 15;   // 512 threads <-> 512 (d,p) pairs
        float n2 = 0.f;
#pragma unroll
        for (int cc = 0; cc < COUT; ++cc) {
          float sv = ldsSV[(cc * DOUT + d) * NP + pp];
          n2 += sv * sv;
        }
        float f = (n2 / (1.0f + n2)) * rsqrtf(n2 + 1e-8f);
#pragma unroll
        for (int cc = 0; cc < COUT; ++cc)
          ldsSV[(cc * DOUT + d) * NP + pp] *= f;   // v in place
      }
      __syncthreads();
    }
  } // sweeps

  // ---- write s (iteration-3) straight from registers, coalesced in w ----
  if (p < NPV) {
    int w = w0 + p;
#pragma unroll
    for (int j = 0; j < 8; ++j) {
      int d0 = j + 8 * hi;
      int d1 = 16 + j + 8 * hi;
      out[((b * COUT + c) * DOUT + d0) * (HOUT * WOUT) + h * WOUT + w] = sacc0[j];
      out[((b * COUT + c) * DOUT + d1) * (HOUT * WOUT) + h * WOUT + w] = sacc1[j];
    }
  }
}

// ---------------------------------------------------------------------------
extern "C" void kernel_launch(void* const* d_in, const int* in_sizes, int n_in,
                              void* d_out, int out_size, void* d_ws, size_t ws_size,
                              hipStream_t stream) {
  const float* x  = (const float*)d_in[0];   // (8,8,32,32,32)
  const float* w  = (const float*)d_in[1];   // (16384,8,3,3)
  const float* pr = (const float*)d_in[2];   // (1,32,32,1,1)
  float* out = (float*)d_out;                // (8,16,32,30,30)
  __bf16* pw = (__bf16*)d_ws;                // 3.1 MB packed bf16 weights

  int total = DIN * 32 * KSTEPS * 32 * 16;   // 1,572,864 packed elements
  prepack_w<<<(total + 255) / 256, 256, 0, stream>>>(w, pw, total);

  capsule_fused<<<dim3(2, HOUT, B_), 512, 0, stream>>>(x, pw, pr, out);
}
